// Qwen2MoeDecoderLayer_13245679140938
// MI455X (gfx1250) — compile-verified, matrix operand order checked by reading, not run on hardware
//
#include <hip/hip_runtime.h>
#include <hip/hip_bf16.h>

// ---------------------------------------------------------------------------
// Qwen2-MoE decoder layer for MI455X (gfx1250, wave32, WMMA).
// All GEMMs run on v_wmma_f32_16x16x32_f16 with fp32 accumulation.
// ---------------------------------------------------------------------------

#define BB   4
#define SS   1024
#define DD   2048
#define NH   16
#define HDD  128
#define HHD  (NH * HDD)       // 2048
#define EE   8
#define FMM  1408
#define FSS  5632
#define NTOK (BB * SS)        // 4096

#define LDSTR 20              // LDS row stride in uints: rows 16B-aligned,
                              // banks 20*r % 64 = multiples of 4 -> b128 conflict-free

typedef __attribute__((ext_vector_type(16))) _Float16     v16h;
typedef __attribute__((ext_vector_type(8)))  float        v8f;
typedef __attribute__((ext_vector_type(8)))  unsigned int v8u;

__device__ __forceinline__ unsigned int pack_f16x2(float lo, float hi)
{
    union { _Float16 h[2]; unsigned int u; } pk;
    pk.h[0] = (_Float16)lo;
    pk.h[1] = (_Float16)hi;
    return pk.u;
}

// One WMMA operand = two contiguous uint4 LDS loads (fragment-ordered layout):
// VGPR j holds K-pair k2(j) = (j<4 ? j : 8+(j-4)) + 4*lh
__device__ __forceinline__ v16h frag_load(const unsigned int* __restrict__ rowPtr, int lh)
{
    uint4 lo = *(const uint4*)(rowPtr + 4 * lh);
    uint4 hi = *(const uint4*)(rowPtr + 8 + 4 * lh);
    v8u u;
    u[0] = lo.x; u[1] = lo.y; u[2] = lo.z; u[3] = lo.w;
    u[4] = hi.x; u[5] = hi.y; u[6] = hi.z; u[7] = hi.w;
    return __builtin_bit_cast(v16h, u);
}

// ---------------------------------------------------------------------------
// Tiled WMMA GEMM:  C[M,N] = alpha * A[M,K] @ B[K,N]  (+ epilogue)
//   TRANSB=1 => B element [k][n] read from Bm[n*ldb + k]
//   Epilogue: if rowScale: C += rowScale[m*rsStride] * v   (MoE accumulate)
//             else        : C  = v + bias[n]? + resid[m*ldc+n]?
//
// Block tile 128(M) x 64(N), K step 32. 8 waves as 4(M) x 2(N); each wave
// owns a 32x32 output = 4 WMMA accumulators fed by 2 A + 2 B fragments.
//
// LDS is K-PAIR PACKED and FRAGMENT-ORDERED: uint at [row][k/2] so each
// fragment is two ds_load_b128 + bit_cast (no 16-bit mov traffic).
//
// REQUIRES: M % 128 == 0, N % 64 == 0, K % 32 == 0, lda/ldb % 4 == 0
// (true for every GEMM in this layer) -> no bounds guards, coalesced
// global_load_b128.
// ---------------------------------------------------------------------------
template <bool TRANSB>
__global__ __launch_bounds__(256)
void k_gemm(const float* __restrict__ A,  int lda,
            const float* __restrict__ Bm, int ldb,
            const float* __restrict__ bias,
            const float* __restrict__ resid,
            const float* __restrict__ rowScale, int rsStride,
            float* __restrict__ C, int ldc,
            int M, int N, int Kd, float alpha)
{
    __shared__ unsigned int Asp[128][LDSTR];  // [m][k2]
    __shared__ unsigned int Bsp[64][LDSTR];   // [n][k2]

    const int tid  = threadIdx.x;
    const int wid  = tid >> 5;
    const int lane = tid & 31;
    const int wm   = wid & 3;          // wave row  (0..3)  -> 32 rows each
    const int wn   = wid >> 2;         // wave col  (0..1)  -> 32 cols each
    const int lh   = lane >> 4;        // lane half (0/1)
    const int mr   = lane & 15;
    const int bm   = blockIdx.y * 128;
    const int bn   = blockIdx.x * 64;

    v8f acc00 = {}, acc01 = {}, acc10 = {}, acc11 = {};

    for (int kt = 0; kt < Kd; kt += 32) {
        // ---- stage A tile (128 x 32): float4 -> two packed f16x2 uints ----
        #pragma unroll
        for (int i = 0; i < 4; ++i) {
            int idx = tid + i * 256;          // 0..1023
            int kq  = idx & 7;                // float4 index within row
            int m   = idx >> 3;               // 0..127
            const float4 v = *(const float4*)(A + (size_t)(bm + m) * lda + kt + 4 * kq);
            uint2 p;
            p.x = pack_f16x2(v.x, v.y);
            p.y = pack_f16x2(v.z, v.w);
            *(uint2*)&Asp[m][2 * kq] = p;
        }
        // ---- stage B tile (32 x 64) ----
        if (TRANSB) {
            #pragma unroll
            for (int i = 0; i < 2; ++i) {
                int idx = tid + i * 256;      // 0..511
                int kq  = idx & 7;
                int n   = idx >> 3;           // 0..63
                const float4 v = *(const float4*)(Bm + (size_t)(bn + n) * ldb + kt + 4 * kq);
                uint2 p;
                p.x = pack_f16x2(v.x, v.y);
                p.y = pack_f16x2(v.z, v.w);
                *(uint2*)&Bsp[n][2 * kq] = p;
            }
        } else {
            #pragma unroll
            for (int i = 0; i < 4; ++i) {
                int idx = tid + i * 256;      // 0..1023
                int n   = idx & 63;           // varies fastest -> coalesced
                int k2  = idx >> 6;           // 0..15
                float lo = Bm[(size_t)(kt + 2 * k2)     * ldb + bn + n];
                float hi = Bm[(size_t)(kt + 2 * k2 + 1) * ldb + bn + n];
                Bsp[n][k2] = pack_f16x2(lo, hi);
            }
        }
        // prefetch next A tile through the gfx1250 prefetch path
        if (kt + 32 < Kd)
            __builtin_prefetch(A + (size_t)(bm + (tid >> 1)) * lda + kt + 32, 0, 1);
        __syncthreads();

        v16h a0 = frag_load(&Asp[wm * 32 + mr][0],      lh);
        v16h a1 = frag_load(&Asp[wm * 32 + 16 + mr][0], lh);
        v16h b0 = frag_load(&Bsp[wn * 32 + mr][0],      lh);
        v16h b1 = frag_load(&Bsp[wn * 32 + 16 + mr][0], lh);

        acc00 = __builtin_amdgcn_wmma_f32_16x16x32_f16(false, a0, false, b0, (short)0, acc00, false, false);
        acc01 = __builtin_amdgcn_wmma_f32_16x16x32_f16(false, a0, false, b1, (short)0, acc01, false, false);
        acc10 = __builtin_amdgcn_wmma_f32_16x16x32_f16(false, a1, false, b0, (short)0, acc10, false, false);
        acc11 = __builtin_amdgcn_wmma_f32_16x16x32_f16(false, a1, false, b1, (short)0, acc11, false, false);
        __syncthreads();
    }

    // ---- epilogue: C/D layout = lane holds col n=lane%16, VGPR r -> row r+8*lh
    #pragma unroll
    for (int r = 0; r < 8; ++r) {
        int gm0 = bm + wm * 32 + r + lh * 8;
        int gm1 = gm0 + 16;
        int gn0 = bn + wn * 32 + mr;
        int gn1 = gn0 + 16;
        float v00 = acc00[r] * alpha, v01 = acc01[r] * alpha;
        float v10 = acc10[r] * alpha, v11 = acc11[r] * alpha;
        size_t o00 = (size_t)gm0 * ldc + gn0, o01 = (size_t)gm0 * ldc + gn1;
        size_t o10 = (size_t)gm1 * ldc + gn0, o11 = (size_t)gm1 * ldc + gn1;
        if (rowScale) {
            float w0 = rowScale[(size_t)gm0 * rsStride];
            float w1 = rowScale[(size_t)gm1 * rsStride];
            C[o00] += w0 * v00;  C[o01] += w0 * v01;
            C[o10] += w1 * v10;  C[o11] += w1 * v11;
        } else {
            if (bias) {
                float b0v = bias[gn0], b1v = bias[gn1];
                v00 += b0v; v01 += b1v; v10 += b0v; v11 += b1v;
            }
            if (resid) {
                v00 += resid[o00]; v01 += resid[o01];
                v10 += resid[o10]; v11 += resid[o11];
            }
            C[o00] = v00;  C[o01] = v01;
            C[o10] = v10;  C[o11] = v11;
        }
    }
}

// ---------------------------------------------------------------------------
// RMSNorm: one row (D elems) per block
// ---------------------------------------------------------------------------
__global__ __launch_bounds__(256)
void k_rmsnorm(const float* __restrict__ x, const float* __restrict__ scale,
               float* __restrict__ out, int D)
{
    const int row = blockIdx.x;
    const float* xr = x + (size_t)row * D;
    float s = 0.0f;
    for (int i = threadIdx.x; i < D; i += 256) { float v = xr[i]; s += v * v; }
    __shared__ float red[256];
    red[threadIdx.x] = s;
    __syncthreads();
    for (int o = 128; o > 0; o >>= 1) {
        if (threadIdx.x < o) red[threadIdx.x] += red[threadIdx.x + o];
        __syncthreads();
    }
    float r = rsqrtf(red[0] / (float)D + 1e-6f);
    for (int i = threadIdx.x; i < D; i += 256)
        out[(size_t)row * D + i] = xr[i] * r * scale[i];
}

// ---------------------------------------------------------------------------
// RoPE applied in-place to q,k laid out [B,S,H,HD]
// ---------------------------------------------------------------------------
__global__ void k_rope(float* __restrict__ q, float* __restrict__ k,
                       const int* __restrict__ pos)
{
    const int half = HDD / 2;
    int idx = blockIdx.x * blockDim.x + threadIdx.x;
    int total = BB * SS * NH * half;
    if (idx >= total) return;
    int i = idx % half;
    int h = (idx / half) % NH;
    int s = (idx / (half * NH)) % SS;
    int b = idx / (half * NH * SS);
    float p   = (float)pos[b * SS + s];
    float inv = __expf(-((2.0f * (float)i) / (float)HDD) * 13.815510557964274f); // ln(1e6)
    float ang = p * inv;
    float c = cosf(ang), sn = sinf(ang);
    size_t base = ((size_t)(b * SS + s) * NH + h) * HDD;
    float q1 = q[base + i], q2 = q[base + half + i];
    q[base + i]        = q1 * c - q2 * sn;
    q[base + half + i] = q2 * c + q1 * sn;
    float k1 = k[base + i], k2 = k[base + half + i];
    k[base + i]        = k1 * c - k2 * sn;
    k[base + half + i] = k2 * c + k1 * sn;
}

// ---------------------------------------------------------------------------
// Causal softmax over one score row (blockIdx.x = z*S + q, z = b*H + h)
// ---------------------------------------------------------------------------
__global__ __launch_bounds__(256)
void k_softmax_causal(float* __restrict__ sc)
{
    const int row = blockIdx.x;
    const int qi  = row % SS;
    float* p = sc + (size_t)row * SS;
    __shared__ float red[256];

    float mx = -1e30f;
    for (int i = threadIdx.x; i <= qi; i += 256) mx = fmaxf(mx, p[i]);
    red[threadIdx.x] = mx;
    __syncthreads();
    for (int o = 128; o > 0; o >>= 1) {
        if (threadIdx.x < o) red[threadIdx.x] = fmaxf(red[threadIdx.x], red[threadIdx.x + o]);
        __syncthreads();
    }
    mx = red[0];
    __syncthreads();

    float s = 0.0f;
    for (int i = threadIdx.x; i <= qi; i += 256) {
        float e = __expf(p[i] - mx);
        p[i] = e;
        s += e;
    }
    red[threadIdx.x] = s;
    __syncthreads();
    for (int o = 128; o > 0; o >>= 1) {
        if (threadIdx.x < o) red[threadIdx.x] += red[threadIdx.x + o];
        __syncthreads();
    }
    float inv = 1.0f / red[0];
    for (int i = threadIdx.x; i <= qi; i += 256) p[i] *= inv;
    for (int i = qi + 1 + threadIdx.x; i < SS; i += 256) p[i] = 0.0f;
}

// ---------------------------------------------------------------------------
// Elementwise helpers
// ---------------------------------------------------------------------------
__global__ void k_silu_mul(const float* __restrict__ g, const float* __restrict__ u,
                           float* __restrict__ out, size_t n)
{
    size_t i = (size_t)blockIdx.x * blockDim.x + threadIdx.x;
    if (i >= n) return;
    float gv = g[i];
    out[i] = (gv / (1.0f + __expf(-gv))) * u[i];
}

__global__ void k_zero(float* __restrict__ p, size_t n)
{
    size_t i = (size_t)blockIdx.x * blockDim.x + threadIdx.x;
    if (i < n) p[i] = 0.0f;
}

// Tiny GEMMs (router N=8, shexp gate N=1): plain per-thread dot products
__global__ void k_small_gemm(const float* __restrict__ A, const float* __restrict__ W,
                             float* __restrict__ C, int M, int N, int Kd)
{
    int idx = blockIdx.x * blockDim.x + threadIdx.x;
    if (idx >= M * N) return;
    int m = idx / N, n = idx % N;
    float s = 0.0f;
    for (int k = 0; k < Kd; ++k) s += A[(size_t)m * Kd + k] * W[(size_t)k * N + n];
    C[idx] = s;
}

// softmax over 8 experts, top-2, renormalized combine weights
__global__ void k_router_combine(const float* __restrict__ logits,
                                 float* __restrict__ comb, int M)
{
    int t = blockIdx.x * blockDim.x + threadIdx.x;
    if (t >= M) return;
    float l[EE];
    float mx = -1e30f;
    for (int e = 0; e < EE; ++e) { l[e] = logits[t * EE + e]; mx = fmaxf(mx, l[e]); }
    float s = 0.0f;
    for (int e = 0; e < EE; ++e) { l[e] = __expf(l[e] - mx); s += l[e]; }
    float inv = 1.0f / s;
    for (int e = 0; e < EE; ++e) l[e] *= inv;
    int i1 = 0; for (int e = 1; e < EE; ++e) if (l[e] > l[i1]) i1 = e;
    float w1 = l[i1]; l[i1] = -1.0f;
    int i2 = 0; for (int e = 1; e < EE; ++e) if (l[e] > l[i2]) i2 = e;
    float w2 = l[i2];
    float tw = w1 + w2;
    for (int e = 0; e < EE; ++e) comb[t * EE + e] = 0.0f;
    comb[t * EE + i1] = w1 / tw;
    comb[t * EE + i2] = w2 / tw;
}

// out = acc + sigmoid(gate_logit[m]) * shared_out
__global__ void k_final(const float* __restrict__ acc, const float* __restrict__ sh,
                        const float* __restrict__ gate, float* __restrict__ out, size_t n)
{
    size_t i = (size_t)blockIdx.x * blockDim.x + threadIdx.x;
    if (i >= n) return;
    size_t m = i / DD;
    float g = 1.0f / (1.0f + __expf(-gate[m]));
    out[i] = acc[i] + g * sh[i];
}

// ---------------------------------------------------------------------------
// Host orchestration
// ---------------------------------------------------------------------------
extern "C" void kernel_launch(void* const* d_in, const int* in_sizes, int n_in,
                              void* d_out, int out_size, void* d_ws, size_t ws_size,
                              hipStream_t stream)
{
    const int*   positions = (const int*)  d_in[0];
    const float* x         = (const float*)d_in[1];
    const float* ln1       = (const float*)d_in[2];
    const float* wq        = (const float*)d_in[3];
    const float* bq        = (const float*)d_in[4];
    const float* wk        = (const float*)d_in[5];
    const float* bk        = (const float*)d_in[6];
    const float* wv        = (const float*)d_in[7];
    const float* bv        = (const float*)d_in[8];
    const float* wo        = (const float*)d_in[9];
    const float* ln2       = (const float*)d_in[10];
    const float* router_w  = (const float*)d_in[11];
    const float* eg_w      = (const float*)d_in[12];   // [E][D][FM]
    const float* eu_w      = (const float*)d_in[13];   // [E][D][FM]
    const float* ed_w      = (const float*)d_in[14];   // [E][FM][D]
    const float* sg_w      = (const float*)d_in[15];   // [D][FS]
    const float* su_w      = (const float*)d_in[16];   // [D][FS]
    const float* sd_w      = (const float*)d_in[17];   // [FS][D]
    const float* shg_w     = (const float*)d_in[18];   // [D][1]
    float* out = (float*)d_out;

    // ---- workspace carve-out ----
    char* wsp = (char*)d_ws;
    auto alloc = [&](size_t nElem) -> float* {
        float* p = (float*)wsp;
        wsp += ((nElem * sizeof(float) + 255) / 256) * 256;
        return p;
    };
    float* hn    = alloc((size_t)NTOK * DD);          // rmsnorm1 out
    float* qf    = alloc((size_t)NTOK * HHD);
    float* kf    = alloc((size_t)NTOK * HHD);
    float* vf    = alloc((size_t)NTOK * HHD);
    float* sc    = alloc((size_t)BB * NH * SS * SS);  // attention scores/probs
    float* of    = alloc((size_t)NTOK * HHD);         // attention output
    float* h2    = alloc((size_t)NTOK * DD);          // o@wo + x
    float* h2n   = alloc((size_t)NTOK * DD);          // rmsnorm2 out
    float* sg    = alloc((size_t)NTOK * FSS);
    float* su    = alloc((size_t)NTOK * FSS);
    float* sact  = alloc((size_t)NTOK * FSS);
    float* shout = alloc((size_t)NTOK * DD);
    float* gatel = alloc((size_t)NTOK);
    float* rlog  = alloc((size_t)NTOK * EE);
    float* comb  = alloc((size_t)NTOK * EE);
    float* egb   = alloc((size_t)NTOK * FMM);
    float* eub   = alloc((size_t)NTOK * FMM);
    float* eact  = alloc((size_t)NTOK * FMM);
    float* accb  = alloc((size_t)NTOK * DD);

    auto gemm = [&](const float* A, int lda, const float* Bm, int ldb, bool transB,
                    const float* bias, const float* resid,
                    const float* rowScale, int rsStride,
                    float* C, int ldc, int M, int N, int Kd, float alpha) {
        dim3 g((unsigned)(N / 64), (unsigned)(M / 128));
        if (transB)
            k_gemm<true ><<<g, 256, 0, stream>>>(A, lda, Bm, ldb, bias, resid,
                                                 rowScale, rsStride, C, ldc, M, N, Kd, alpha);
        else
            k_gemm<false><<<g, 256, 0, stream>>>(A, lda, Bm, ldb, bias, resid,
                                                 rowScale, rsStride, C, ldc, M, N, Kd, alpha);
    };

    // 1) rmsnorm1
    k_rmsnorm<<<NTOK, 256, 0, stream>>>(x, ln1, hn, DD);

    // 2) QKV projections (with bias)
    gemm(hn, DD, wq, HHD, false, bq, nullptr, nullptr, 0, qf, HHD, NTOK, HHD, DD, 1.0f);
    gemm(hn, DD, wk, HHD, false, bk, nullptr, nullptr, 0, kf, HHD, NTOK, HHD, DD, 1.0f);
    gemm(hn, DD, wv, HHD, false, bv, nullptr, nullptr, 0, vf, HHD, NTOK, HHD, DD, 1.0f);

    // 3) RoPE (in place; qf/kf fully rewritten every call by the GEMMs above)
    {
        int total = BB * SS * NH * (HDD / 2);
        k_rope<<<(total + 255) / 256, 256, 0, stream>>>(qf, kf, positions);
    }

    // 4) scores = (Q @ K^T) * HD^-0.5 per (b,h)
    const float scl = 0.08838834764831845f;   // 1/sqrt(128)
    for (int b = 0; b < BB; ++b)
        for (int h = 0; h < NH; ++h) {
            size_t off = (size_t)b * SS * HHD + (size_t)h * HDD;
            int z = b * NH + h;
            gemm(qf + off, HHD, kf + off, HHD, /*transB=*/true,
                 nullptr, nullptr, nullptr, 0,
                 sc + (size_t)z * SS * SS, SS, SS, SS, HDD, scl);
        }

    // 5) causal softmax
    k_softmax_causal<<<BB * NH * SS, 256, 0, stream>>>(sc);

    // 6) O = P @ V per (b,h)
    for (int b = 0; b < BB; ++b)
        for (int h = 0; h < NH; ++h) {
            size_t off = (size_t)b * SS * HHD + (size_t)h * HDD;
            int z = b * NH + h;
            gemm(sc + (size_t)z * SS * SS, SS, vf + off, HHD, false,
                 nullptr, nullptr, nullptr, 0,
                 of + off, HHD, SS, HDD, SS, 1.0f);
        }

    // 7) h2 = O @ wo + x   (residual fused into GEMM epilogue)
    gemm(of, HHD, wo, DD, false, nullptr, x, nullptr, 0, h2, DD, NTOK, DD, HHD, 1.0f);

    // 8) rmsnorm2
    k_rmsnorm<<<NTOK, 256, 0, stream>>>(h2, ln2, h2n, DD);

    // 9) shared expert: silu(t@sg_w) * (t@su_w) @ sd_w
    gemm(h2n, DD, sg_w, FSS, false, nullptr, nullptr, nullptr, 0, sg, FSS, NTOK, FSS, DD, 1.0f);
    gemm(h2n, DD, su_w, FSS, false, nullptr, nullptr, nullptr, 0, su, FSS, NTOK, FSS, DD, 1.0f);
    {
        size_t n = (size_t)NTOK * FSS;
        k_silu_mul<<<(unsigned)((n + 255) / 256), 256, 0, stream>>>(sg, su, sact, n);
    }
    gemm(sact, FSS, sd_w, DD, false, nullptr, nullptr, nullptr, 0, shout, DD, NTOK, DD, FSS, 1.0f);

    // 10) shexp gate + router (tiny N -> scalar dot-product kernels)
    k_small_gemm<<<(NTOK + 255) / 256, 256, 0, stream>>>(h2n, shg_w, gatel, NTOK, 1, DD);
    k_small_gemm<<<(NTOK * EE + 255) / 256, 256, 0, stream>>>(h2n, router_w, rlog, NTOK, EE, DD);
    k_router_combine<<<(NTOK + 255) / 256, 256, 0, stream>>>(rlog, comb, NTOK);

    // 11) MoE experts, combine-weighted accumulate in the down-proj epilogue
    {
        size_t n = (size_t)NTOK * DD;
        k_zero<<<(unsigned)((n + 255) / 256), 256, 0, stream>>>(accb, n);
    }
    for (int e = 0; e < EE; ++e) {
        const float* egw = eg_w + (size_t)e * DD * FMM;
        const float* euw = eu_w + (size_t)e * DD * FMM;
        const float* edw = ed_w + (size_t)e * FMM * DD;
        gemm(h2n, DD, egw, FMM, false, nullptr, nullptr, nullptr, 0, egb, FMM, NTOK, FMM, DD, 1.0f);
        gemm(h2n, DD, euw, FMM, false, nullptr, nullptr, nullptr, 0, eub, FMM, NTOK, FMM, DD, 1.0f);
        {
            size_t n = (size_t)NTOK * FMM;
            k_silu_mul<<<(unsigned)((n + 255) / 256), 256, 0, stream>>>(egb, eub, eact, n);
        }
        // acc += combine[:, e] * (act @ ed_w[e])
        gemm(eact, FMM, edw, DD, false, nullptr, nullptr, comb + e, EE,
             accb, DD, NTOK, DD, FMM, 1.0f);
    }

    // 12) out = acc + sigmoid(gate) * shared_expert_out
    {
        size_t n = (size_t)NTOK * DD;
        k_final<<<(unsigned)((n + 255) / 256), 256, 0, stream>>>(accb, shout, gatel, out, n);
    }
    (void)in_sizes; (void)n_in; (void)out_size; (void)ws_size;
}
